// GRUModel_6365141533330
// MI455X (gfx1250) — matrix operand order repaired and checked by reading, not compiled, not run on hardware
//
#include <hip/hip_runtime.h>
#include <math.h>

// GRU: B=256, T=1024, D=128, H=64, classes=2.
// Phase 1: x_gates = x @ W_ih^T + b_ih  via v_wmma_f32_16x16x32_f16, stored f16 in d_ws
//          (layout (t, b, 3H), 1024*256*192*2 = 96 MB <= ws_size assumed).
// Phase 2: 1024-step scan, 16 WGs x 4 waves; per wave 6 WMMA/step, W_hh fragments
//          register-resident; per-step 6KB x_gates slab staged into a ping-pong LDS
//          buffer with GLOBAL_LOAD_ASYNC_TO_LDS_B128 (ASYNCcnt), drained with
//          s_wait_asynccnt right before the single per-step barrier.

#define BATCH 256
#define SEQT  1024
#define DIN   128
#define HID   64
#define GATES 192
#define NCLS  2

typedef __attribute__((ext_vector_type(16))) _Float16 v16h;
typedef __attribute__((ext_vector_type(8)))  _Float16 v8h;
typedef __attribute__((ext_vector_type(8)))  float    v8f;
typedef __attribute__((ext_vector_type(4)))  float    v4f;

union V16H { v16h v; v8h h2[2]; _Float16 e[16]; };

// LDS byte offset of a generic pointer that is known to point into LDS.
static __device__ __forceinline__ unsigned lds_off(const void* p) {
    return (unsigned)(unsigned long long)
        (__attribute__((address_space(3))) const void*)p;
}

// Async 16-byte global -> LDS copy (tracked by ASYNCcnt).
static __device__ __forceinline__ void async_copy_b128(unsigned lds_byte,
                                                       const void* gaddr) {
    asm volatile("global_load_async_to_lds_b128 %0, %1, off"
                 :
                 : "v"(lds_byte), "v"((unsigned long long)gaddr)
                 : "memory");
}

static __device__ __forceinline__ void wait_asynccnt0() {
    asm volatile("s_wait_asynccnt 0x0" ::: "memory");
}

// A-fragment (16x32 f16). Lane provides row M=l%16. K pattern per ISA:
//   i=0..7  -> K = 8*(l>>4) + i ; i=8..15 -> K = 16 + 8*(l>>4) + (i-8).
// p must point at (row base + kt*32 + 8*(l>>4)) of an f32 source row.
static __device__ __forceinline__ v16h load_frag_a_f32(const float* __restrict__ p) {
    V16H a;
    v4f r0 = *(const v4f*)(p);      // k +0..3
    v4f r1 = *(const v4f*)(p + 4);  // k +4..7
    v4f r2 = *(const v4f*)(p + 16); // k +16..19
    v4f r3 = *(const v4f*)(p + 20); // k +20..23
#pragma unroll
    for (int i = 0; i < 4; ++i) {
        a.e[i]      = (_Float16)r0[i];
        a.e[4 + i]  = (_Float16)r1[i];
        a.e[8 + i]  = (_Float16)r2[i];
        a.e[12 + i] = (_Float16)r3[i];
    }
    return a.v;
}

// B-fragment (32x16 f16). Lane provides col N=l%16; K = 8*(l>>4) + i (i<8), 16 + 8*(l>>4) + (i-8) (i>=8).
static __device__ __forceinline__ v16h load_frag_b_f32(const float* __restrict__ p) {
    V16H b;
    v4f r0 = *(const v4f*)(p);
    v4f r1 = *(const v4f*)(p + 4);
    v4f r2 = *(const v4f*)(p + 16);
    v4f r3 = *(const v4f*)(p + 20);
#pragma unroll
    for (int i = 0; i < 4; ++i) {
        b.e[i]      = (_Float16)r0[i];
        b.e[4 + i]  = (_Float16)r1[i];
        b.e[8 + i]  = (_Float16)r2[i];
        b.e[12 + i] = (_Float16)r3[i];
    }
    return b.v;
}

static __device__ __forceinline__ v8f wmma_f16(v16h a, v16h b, v8f c) {
    return __builtin_amdgcn_wmma_f32_16x16x32_f16(false, a, false, b, (short)0, c,
                                                  false, false);
}

// ---------------------------------------------------------------- Phase 1
__global__ __launch_bounds__(128) void gru_xgates_kernel(
    const float* __restrict__ x, const float* __restrict__ Wih,
    const float* __restrict__ bih, _Float16* __restrict__ xg) {
    const int lane = threadIdx.x & 31;
    const int wv   = threadIdx.x >> 5;  // wave 0..3 owns N-tiles wv*3 .. wv*3+2
    const int lg   = lane >> 4;
    const int ln   = lane & 15;

    v16h Bf[3][4];
    float bias[3];
#pragma unroll
    for (int nt = 0; nt < 3; ++nt) {
        const int g = (wv * 3 + nt) * 16 + ln;  // W_ih row (gate output index)
        bias[nt] = bih[g];
#pragma unroll
        for (int kt = 0; kt < 4; ++kt)
            Bf[nt][kt] = load_frag_b_f32(Wih + (size_t)g * DIN + kt * 32 + 8 * lg);
    }

    const int tiles = (BATCH * SEQT) / 16;  // 16384 row tiles
    for (int tile = blockIdx.x; tile < tiles; tile += gridDim.x) {
        const int rbase = tile * 16;
        const int b  = rbase >> 10;         // x row = b*T + t
        const int t0 = rbase & (SEQT - 1);  // 16 rows share b, t = t0..t0+15

        const float* xrow = x + (size_t)(rbase + ln) * DIN + 8 * lg;
        v16h A[4];
#pragma unroll
        for (int kt = 0; kt < 4; ++kt) A[kt] = load_frag_a_f32(xrow + kt * 32);

#pragma unroll
        for (int nt = 0; nt < 3; ++nt) {
            v8f acc;
#pragma unroll
            for (int j = 0; j < 8; ++j) acc[j] = bias[nt];
#pragma unroll
            for (int kt = 0; kt < 4; ++kt) acc = wmma_f16(A[kt], Bf[nt][kt], acc);
            const int col = (wv * 3 + nt) * 16 + ln;
#pragma unroll
            for (int j = 0; j < 8; ++j) {
                const int m = j + 8 * lg;  // D row
                xg[((size_t)(t0 + m) * BATCH + b) * GATES + col] = (_Float16)acc[j];
            }
        }
    }
}

// ---------------------------------------------------------------- Phase 2
__global__ __launch_bounds__(128) void gru_scan_kernel(
    const _Float16* __restrict__ xg, const float* __restrict__ Whh,
    const float* __restrict__ bhh, const float* __restrict__ W1,
    const float* __restrict__ b1, const float* __restrict__ W2,
    const float* __restrict__ b2, float* __restrict__ out) {
    __shared__ __align__(16) _Float16 hbuf[2][16][HID];    // ping-pong h (f16)
    __shared__ __align__(16) _Float16 xbuf[2][16][GATES];  // ping-pong x_gates slab
    __shared__ float hfin[16][HID];

    const int lane = threadIdx.x & 31;
    const int wv   = threadIdx.x >> 5;  // wave owns h columns wv*16 .. wv*16+15
    const int lg   = lane >> 4;
    const int ln   = lane & 15;
    const int b0   = blockIdx.x * 16;   // batch rows b0 .. b0+15
    const int col  = wv * 16 + ln;      // this lane's output column (0..63)

    // Register-resident recurrent weight fragments: gate {r,z,n} x ktile {0,1}.
    // B[k][n] = W_hg[n][k]  (h @ W_hg^T), row n = g*64+col, k runs 8*lg+0..7 and 16+8*lg+0..7.
    v16h Bf[3][2];
#pragma unroll
    for (int g = 0; g < 3; ++g)
#pragma unroll
        for (int kt = 0; kt < 2; ++kt)
            Bf[g][kt] =
                load_frag_b_f32(Whh + (size_t)(g * HID + col) * HID + kt * 32 + 8 * lg);

    const float bhr = bhh[col], bhz = bhh[HID + col], bhn = bhh[2 * HID + col];

    v16h A0 = {}, A1 = {};  // h(0) = 0
    float hval[8];
#pragma unroll
    for (int j = 0; j < 8; ++j) hval[j] = 0.f;

    // Prologue: stage slab t=0 into xbuf[0] (3 async 16B chunks per thread).
    {
        const _Float16* gsrc = xg + ((size_t)0 * BATCH + b0) * GATES;
        const unsigned   ldb = lds_off(&xbuf[0][0][0]);
#pragma unroll
        for (int i = 0; i < 3; ++i) {
            const int c = threadIdx.x + 128 * i;  // 384 chunks of 16B = 6 KB
            async_copy_b128(ldb + c * 16, gsrc + c * 8);
        }
        wait_asynccnt0();
        __syncthreads();
    }

    for (int t = 0; t < SEQT; ++t) {
        const int p = t & 1;

        // Gate pre-activations from the LDS-staged slab (+ recurrent biases).
        v8f accR, accZ, accN;
        float xn[8];
#pragma unroll
        for (int j = 0; j < 8; ++j) {
            const int m = j + 8 * lg;
            accR[j] = bhr + (float)xbuf[p][m][col];            // xr + b_hr
            accZ[j] = bhz + (float)xbuf[p][m][HID + col];      // xz + b_hz
            xn[j]   = (float)xbuf[p][m][2 * HID + col];        // xn kept aside
            accN[j] = bhn;                                     // b_hn inside r*(...)
        }

        // Kick off async staging of slab t+1 into the other buffer; it overlaps
        // with the WMMAs + gate math below.
        if (t + 1 < SEQT) {
            const _Float16* gsrc = xg + ((size_t)(t + 1) * BATCH + b0) * GATES;
            const unsigned   ldb = lds_off(&xbuf[p ^ 1][0][0]);
#pragma unroll
            for (int i = 0; i < 3; ++i) {
                const int c = threadIdx.x + 128 * i;
                async_copy_b128(ldb + c * 16, gsrc + c * 8);
            }
        }

        accR = wmma_f16(A0, Bf[0][0], accR);
        accR = wmma_f16(A1, Bf[0][1], accR);
        accZ = wmma_f16(A0, Bf[1][0], accZ);
        accZ = wmma_f16(A1, Bf[1][1], accZ);
        accN = wmma_f16(A0, Bf[2][0], accN);
        accN = wmma_f16(A1, Bf[2][1], accN);

#pragma unroll
        for (int j = 0; j < 8; ++j) {
            const int m = j + 8 * lg;
            const float r  = 1.f / (1.f + __expf(-accR[j]));
            const float z  = 1.f / (1.f + __expf(-accZ[j]));
            const float e2 = __expf(2.f * (xn[j] + r * accN[j]));
            const float n  = (e2 - 1.f) / (e2 + 1.f);  // tanh
            const float hn = (1.f - z) * n + z * hval[j];
            hval[j] = hn;
            hbuf[p][m][col] = (_Float16)hn;
        }

        wait_asynccnt0();  // own chunks of slab t+1 are in LDS
        __syncthreads();   // all waves' chunks + h(t+1) visible; one barrier/step

        // Rebuild A-fragments of h(t+1): two aligned 16B LDS loads per fragment.
        const int kb = 8 * lg;
        V16H a;
        a.h2[0] = *(const v8h*)&hbuf[p][ln][kb];
        a.h2[1] = *(const v8h*)&hbuf[p][ln][kb + 16];
        A0 = a.v;
        a.h2[0] = *(const v8h*)&hbuf[p][ln][32 + kb];
        a.h2[1] = *(const v8h*)&hbuf[p][ln][32 + kb + 16];
        A1 = a.v;
    }

    // ---- classifier head on h(T) (f32 master copy lives in hval) ----
#pragma unroll
    for (int j = 0; j < 8; ++j) hfin[j + 8 * lg][col] = hval[j];
    __syncthreads();

    if (threadIdx.x < 16) {
        const int m = threadIdx.x;
        float l0 = b2[0], l1 = b2[1];
        for (int jj = 0; jj < HID; ++jj) {
            float s = b1[jj];
#pragma unroll 8
            for (int k = 0; k < HID; ++k) s += hfin[m][k] * W1[jj * HID + k];
            l0 += W2[jj] * s;
            l1 += W2[HID + jj] * s;
        }
        const float mx  = fmaxf(l0, l1);
        const float lse = mx + __logf(__expf(l0 - mx) + __expf(l1 - mx));
        out[(size_t)(b0 + m) * NCLS + 0] = l0 - lse;
        out[(size_t)(b0 + m) * NCLS + 1] = l1 - lse;
    }
}

// ---------------------------------------------------------------- launch
extern "C" void kernel_launch(void* const* d_in, const int* in_sizes, int n_in,
                              void* d_out, int out_size, void* d_ws, size_t ws_size,
                              hipStream_t stream) {
    const float* x   = (const float*)d_in[0];
    const float* Wih = (const float*)d_in[1];
    const float* bih = (const float*)d_in[2];
    const float* Whh = (const float*)d_in[3];
    const float* bhh = (const float*)d_in[4];
    const float* W1  = (const float*)d_in[5];
    const float* b1  = (const float*)d_in[6];
    const float* W2  = (const float*)d_in[7];
    const float* b2  = (const float*)d_in[8];
    float* out = (float*)d_out;

    _Float16* xg = (_Float16*)d_ws;  // needs 1024*256*192*2 = 96 MB

    gru_xgates_kernel<<<512, 128, 0, stream>>>(x, Wih, bih, xg);
    gru_scan_kernel<<<BATCH / 16, 128, 0, stream>>>(xg, Whh, bhh, W1, b1, W2, b2, out);
}
